// CRF_80573586473282
// MI455X (gfx1250) — matrix operand order repaired and checked by reading, not compiled, not run on hardware
//
#include <hip/hip_runtime.h>
#include <hip/hip_bf16.h>

#define NT   128   // NUM_TAGS
#define SL   512   // SEQ_LEN
#define NB   256   // BATCH
#define ASTR 132   // alpha LDS row stride (floats), padded vs 128 to dodge bank conflicts
#define PSTR 136   // p (f16) LDS row stride (halves): 272B rows, 16B aligned, conflict-free b128 loads

typedef __attribute__((ext_vector_type(16))) _Float16 v16h;
typedef __attribute__((ext_vector_type(8)))  _Float16 v8h;
typedef __attribute__((ext_vector_type(8)))  float    v8f;

__device__ __forceinline__ float fast_exp(float x) {
#if __has_builtin(__builtin_amdgcn_exp2f)
    return __builtin_amdgcn_exp2f(x * 1.44269504089f);   // bare v_exp_f32
#else
    return __expf(x);
#endif
}

__device__ __forceinline__ float fast_log(float x) {
#if __has_builtin(__builtin_amdgcn_logf)
    return __builtin_amdgcn_logf(x) * 0.69314718056f;    // bare v_log_f32
#else
    return __logf(x);
#endif
}

// branchless bit-select: bit ? a : b  (no control flow possible)
__device__ __forceinline__ float bitsel(unsigned bit, float a, float b) {
    unsigned sel = 0u - (bit & 1u);                      // 0x0 or 0xFFFFFFFF
    return __uint_as_float((__float_as_uint(a) & sel) |
                           (__float_as_uint(b) & ~sel));
}

// ---------------------------------------------------------------------------
// Kernel 1: E = exp(transitions), stored f16 row-major [k][n]
// ---------------------------------------------------------------------------
__global__ void crf_exp_trans(const float* __restrict__ trans,
                              _Float16* __restrict__ Eh) {
    int i = blockIdx.x * 256 + threadIdx.x;
    if (i < NT * NT) Eh[i] = (_Float16)__expf(trans[i]);
}

// ---------------------------------------------------------------------------
// Kernel 2: gold-path numerator per batch (gathers + sequential scan)
// ---------------------------------------------------------------------------
__global__ __launch_bounds__(256)
void crf_numerator(const float* __restrict__ em,
                   const int* __restrict__ tags,
                   const unsigned char* __restrict__ mask,
                   const float* __restrict__ startT,
                   const float* __restrict__ endT,
                   const float* __restrict__ trans,
                   float* __restrict__ numer) {
    int b = blockIdx.x * blockDim.x + threadIdx.x;
    if (b >= NB) return;
    const int*           tg = tags + b * SL;
    const unsigned char* mk = mask + b * SL;
    int prev = tg[0];
    float s  = startT[prev] + em[(b * SL + 0) * NT + prev];
    int cnt  = mk[0] ? 1 : 0;
    for (int t = 1; t < SL; ++t) {
        int cur = tg[t];
        if (mk[t]) {
            s += em[(b * SL + t) * NT + cur] + trans[prev * NT + cur];
            ++cnt;
        }
        prev = cur;
    }
    int lastIdx = cnt > 0 ? cnt - 1 : 0;
    s += endT[tg[lastIdx]];
    numer[b] = s;
}

// ---------------------------------------------------------------------------
// Kernel 3: forward algorithm via WMMA.  One block = 16 batches, 8 waves.
// Wave w owns output-tag columns [16w, 16w+16).  B fragment (E) is register-
// resident for the whole sequence; each lane also keeps its 8 alpha values
// (D-layout) in registers, mirrored to LDS for the cross-thread max / p
// phases.  A fragment (p = exp(alpha - rowmax), f16) rebuilt in LDS each
// step; C accumulates K=128 with 4 chained WMMAs.  Masked update is a pure
// bitwise select on registers -> no exec-mask branches in the hot loop.
// ---------------------------------------------------------------------------
__global__ __launch_bounds__(256)
void crf_forward(const float* __restrict__ em,
                 const unsigned char* __restrict__ mask,
                 const float* __restrict__ startT,
                 const float* __restrict__ endT,
                 const _Float16* __restrict__ Eh,
                 const float* __restrict__ numer,
                 float* __restrict__ out) {
    __shared__ float                  alpha[16 * ASTR];
    __shared__ __align__(16) _Float16 pH[16 * PSTR];
    __shared__ float                  pred[16 * 16];
    __shared__ __align__(32) float    rowM[16];
    __shared__ unsigned               mbitsS;

    const int tid  = threadIdx.x;
    const int lane = tid & 31;
    const int wv   = tid >> 5;          // wave id 0..7 -> tag columns 16*wv..
    const int b0   = blockIdx.x * 16;   // first batch of this tile
    const int r    = tid & 15;          // row (batch-in-tile) for 256-thread phases
    const int seg  = tid >> 4;          // 8-column segment 0..15

    // --- persistent B fragments: E[k][n], n = 16*wv + (lane&15),
    //     lane<16 covers K kbase+0..15, lane>=16 covers K kbase+16..31 ----
    const int bn = wv * 16 + (lane & 15);
    const int kb = (lane & 16);         // 0 or 16
    v16h Bfrag[4];
#pragma unroll
    for (int c = 0; c < 4; ++c) {
        v16h bv;
        int k0 = c * 32 + kb;
#pragma unroll
        for (int h = 0; h < 16; ++h)
            bv[h] = Eh[(k0 + h) * NT + bn];
        Bfrag[c] = bv;
    }

    // --- alpha0 = start + emissions[:,0,:] ---
#pragma unroll
    for (int c = 0; c < 8; ++c) {
        int n = seg * 8 + c;
        alpha[r * ASTR + n] = startT[n] + em[((b0 + r) * SL + 0) * NT + n];
    }
    __syncthreads();

    const int row16 = lane & 15;        // A-fragment row
    const int hiK   = (lane >> 4) & 1;  // A-fragment lane-half K offset
    const int dN    = wv * 16 + (lane & 15);       // D column (tag)
    const int dR0   = (lane & 16) ? 8 : 0;         // D first row

    // lane-resident alpha (D layout): av[v] = alpha[dR0+v][dN]
    float av[8];
#pragma unroll
    for (int v = 0; v < 8; ++v)
        av[v] = alpha[(dR0 + v) * ASTR + dN];

    for (int t = 1; t < SL; ++t) {
        // prefetch next step's emissions tile (speculative; silently dropped OOB)
        __builtin_prefetch(&em[((b0 + r) * SL + t + 1) * NT + seg * 8], 0, 0);

        // 1) partial row max of alpha
        float mx = alpha[r * ASTR + seg * 8];
#pragma unroll
        for (int c = 1; c < 8; ++c)
            mx = fmaxf(mx, alpha[r * ASTR + seg * 8 + c]);
        pred[r * 16 + seg] = mx;
        __syncthreads();

        // 2) reduce row max (threads 0..15); wave 0 ballots the step mask
        if (wv == 0) {
            bool m = mask[(b0 + (lane & 15)) * SL + t] != 0;
            unsigned bal = __builtin_amdgcn_ballot_w32(m);
            if (lane == 0) mbitsS = bal & 0xFFFFu;
        }
        if (tid < 16) {
            float mm = pred[tid * 16];
#pragma unroll
            for (int s2 = 1; s2 < 16; ++s2)
                mm = fmaxf(mm, pred[tid * 16 + s2]);
            rowM[tid] = mm;
        }
        __syncthreads();

        // 3) p = exp(alpha - rowmax)  -> f16 LDS (A-matrix staging)
        {
            float mm = rowM[r];
#pragma unroll
            for (int c = 0; c < 8; ++c) {
                int n = seg * 8 + c;
                pH[r * PSTR + n] = (_Float16)fast_exp(alpha[r * ASTR + n] - mm);
            }
        }
        __syncthreads();

        // hoisted loads for step 5: emissions, rowM vector, mask bits
        float ev[8];
#pragma unroll
        for (int v = 0; v < 8; ++v)
            ev[v] = em[((b0 + dR0 + v) * SL + t) * NT + dN];
        v8f rmv = *(const v8f*)&rowM[dR0];
        unsigned mb = mbitsS;

        // 4) S = P @ E   (16x128 @ 128x16 per wave, K split into 4x32)
        v8f Cacc = {};
#pragma unroll
        for (int c = 0; c < 4; ++c) {
            int kk = c * 32;
            const v8h lo = *(const v8h*)&pH[row16 * PSTR + kk + hiK * 16];
            const v8h hi = *(const v8h*)&pH[row16 * PSTR + kk + hiK * 16 + 8];
            v16h A = __builtin_shufflevector(lo, hi,
                     0, 1, 2, 3, 4, 5, 6, 7, 8, 9, 10, 11, 12, 13, 14, 15);
            Cacc = __builtin_amdgcn_wmma_f32_16x16x32_f16(
                false, A, false, Bfrag[c], (short)0, Cacc, false, false);
        }

        // 5) alpha' = em_t + rowmax + log(S); bitwise masked select on regs,
        //    unconditional mirror to LDS (straight-line, no branches)
#pragma unroll
        for (int v = 0; v < 8; ++v) {
            int rr = dR0 + v;
            float nv = ev[v] + rmv[v] + fast_log(Cacc[v]);
            av[v] = bitsel(mb >> rr, nv, av[v]);
            alpha[rr * ASTR + dN] = av[v];
        }
        __syncthreads();
    }

    // --- denominator: logsumexp(alpha + end) per row; accumulate llh ---
    {
        float mx = -3.0e38f;
#pragma unroll
        for (int c = 0; c < 8; ++c) {
            int n = seg * 8 + c;
            mx = fmaxf(mx, alpha[r * ASTR + n] + endT[n]);
        }
        pred[r * 16 + seg] = mx;
        __syncthreads();
        if (tid < 16) {
            float mm = pred[tid * 16];
#pragma unroll
            for (int s2 = 1; s2 < 16; ++s2)
                mm = fmaxf(mm, pred[tid * 16 + s2]);
            rowM[tid] = mm;
        }
        __syncthreads();
        float sm = 0.f;
        {
            float mm = rowM[r];
#pragma unroll
            for (int c = 0; c < 8; ++c) {
                int n = seg * 8 + c;
                sm += fast_exp(alpha[r * ASTR + n] + endT[n] - mm);
            }
        }
        __syncthreads();           // pred reuse
        pred[r * 16 + seg] = sm;
        __syncthreads();
        if (tid < 16) {
            float ss = 0.f;
#pragma unroll
            for (int s2 = 0; s2 < 16; ++s2)
                ss += pred[tid * 16 + s2];
            float denom = rowM[tid] + fast_log(ss);
            atomicAdd(out, numer[b0 + tid] - denom);
        }
    }
}

// ---------------------------------------------------------------------------
extern "C" void kernel_launch(void* const* d_in, const int* in_sizes, int n_in,
                              void* d_out, int out_size, void* d_ws, size_t ws_size,
                              hipStream_t stream) {
    const float*         emissions = (const float*)d_in[0];
    const int*           tags      = (const int*)d_in[1];
    const unsigned char* mask      = (const unsigned char*)d_in[2];
    const float*         startT    = (const float*)d_in[3];
    const float*         endT      = (const float*)d_in[4];
    const float*         trans     = (const float*)d_in[5];

    _Float16* Eh    = (_Float16*)d_ws;                       // 128*128*2 = 32 KB
    float*    numer = (float*)((char*)d_ws + NT * NT * sizeof(_Float16)); // 1 KB
    float*    out   = (float*)d_out;

    hipMemsetAsync(out, 0, sizeof(float), stream);

    crf_exp_trans<<<(NT * NT + 255) / 256, 256, 0, stream>>>(trans, Eh);
    crf_numerator<<<1, 256, 0, stream>>>(emissions, tags, mask, startT, endT,
                                         trans, numer);
    crf_forward<<<NB / 16, 256, 0, stream>>>(emissions, mask, startT, endT,
                                             Eh, numer, out);
}
